// MoEClassifier_12635793784890
// MI455X (gfx1250) — matrix operand order repaired and checked by reading, not compiled, not run on hardware
//
#include <hip/hip_runtime.h>
#include <hip/hip_bf16.h>

typedef __attribute__((ext_vector_type(16))) _Float16 v16h;
typedef __attribute__((ext_vector_type(8)))  _Float16 v8h;
typedef __attribute__((ext_vector_type(8)))  float    v8f;

#define NWAVES 8

// ---- LDS byte offsets (all 16B aligned) ----
#define OFF_W1T    0          // f16 [64][104]  bb_w1^T, K=96 pad
#define OFF_W2T    13312      // f16 [64][72]   bb_w2^T
#define OFF_GUT    22528      // f16 [128][72]  gU reshaped [64][128]^T
#define OFF_GVT    40960      // f16 [128][40]  gV reshaped [16->32][128]^T
#define OFF_EW1T   51200      // f16 [16][64][72]  e_w1^T per expert
#define OFF_EW2T   198656     // f16 [16][16][72]  e_w2^T per expert (NCLS pad 16)
#define OFF_BB_B1  235520
#define OFF_BB_B2  235776
#define OFF_BB_G   236032
#define OFF_BB_BE  236288
#define OFF_GB     236544
#define OFF_EB1    236608
#define OFF_EG     240704
#define OFF_EBE    244800
#define OFF_EB2    248896     // f32 [16][16] padded
#define OFF_WAVE   249920
#define WAVE_BYTES 8192       // xa/ua+gbuf arena(3072) | hst(2048) | zst(2048) | wbuf(1024)
#define LDS_TOTAL  (OFF_WAVE + NWAVES * WAVE_BYTES)   // 315456 B <= 320KB

// Branchless exact-GELU: erf via Abramowitz-Stegun 7.1.26 (|err| < 1.5e-7).
__device__ __forceinline__ float gelu_f(float x) {
  float z    = __builtin_fabsf(x) * 0.70710678118654752440f;
  float t    = __builtin_amdgcn_rcpf(__builtin_fmaf(0.3275911f, z, 1.0f));
  float poly = t * __builtin_fmaf(t,
                 __builtin_fmaf(t,
                   __builtin_fmaf(t,
                     __builtin_fmaf(t, 1.061405429f, -1.453152027f),
                     1.421413741f),
                   -0.284496736f),
                 0.254829592f);
  float erfz = __builtin_fmaf(-poly, __expf(-z * z), 1.0f);
  float s    = __builtin_copysignf(erfz, x);
  return 0.5f * x * (1.0f + s);
}

// A fragment: 16x32 f16; lane r=lane&15 -> row M; elems i<8 -> K=8*half+i, i>=8 -> K=16+8*half+(i-8)
__device__ __forceinline__ v16h ldfragA(const _Float16* base, int pitch, int koff) {
  const int lane = threadIdx.x & 31;
  const _Float16* p = base + (lane & 15) * pitch + koff + ((lane >> 4) << 3);
  v8h lo = *(const v8h*)(p);
  v8h hi = *(const v8h*)(p + 16);
  return __builtin_shufflevector(lo, hi, 0,1,2,3,4,5,6,7,8,9,10,11,12,13,14,15);
}

// B fragment: 32x16 f16 (stored transposed [N][K]); lane n=lane&15 -> col N; elems i<8 -> K=8*half+i, i>=8 -> K=16+8*half+(i-8)
__device__ __forceinline__ v16h ldfragB(const _Float16* base, int pitch, int koff) {
  const int lane = threadIdx.x & 31;
  const _Float16* p = base + (lane & 15) * pitch + koff + ((lane >> 4) << 3);
  v8h lo = *(const v8h*)(p);
  v8h hi = *(const v8h*)(p + 16);
  return __builtin_shufflevector(lo, hi, 0,1,2,3,4,5,6,7,8,9,10,11,12,13,14,15);
}

#define WMMA_F16(a,b,c) \
  __builtin_amdgcn_wmma_f32_16x16x32_f16(false,(a),false,(b),(short)0,(c),false,false)

__device__ __forceinline__ void vzero(v8f& a) {
#pragma unroll
  for (int i = 0; i < 8; ++i) a[i] = 0.f;
}

// Register-resident LayerNorm over 64 cols (4 N-tiles x 16 lanes) -> f16 stage[16][64]
__device__ __forceinline__ void ln_store16(v8f acc[4], const float* gma, const float* bta,
                                           _Float16* stage) {
  const int lane = threadIdx.x & 31;
  const int half = lane >> 4, col = lane & 15;
#pragma unroll
  for (int v = 0; v < 8; ++v) {
    float s = acc[0][v] + acc[1][v] + acc[2][v] + acc[3][v];
    float q = acc[0][v]*acc[0][v] + acc[1][v]*acc[1][v]
            + acc[2][v]*acc[2][v] + acc[3][v]*acc[3][v];
#pragma unroll
    for (int m = 1; m < 16; m <<= 1) { s += __shfl_xor(s, m, 32); q += __shfl_xor(q, m, 32); }
    float mean = s * 0.015625f;
    float var  = q * 0.015625f - mean * mean;
    float rstd = rsqrtf(var + 1e-5f);
    const int row = v + 8 * half;
#pragma unroll
    for (int t = 0; t < 4; ++t) {
      int c = t * 16 + col;
      stage[row * 64 + c] = (_Float16)((acc[t][v] - mean) * rstd * gma[c] + bta[c]);
    }
  }
}

__global__ __launch_bounds__(256) void moe_fused_kernel(
    const float* __restrict__ x, const int* __restrict__ uid,
    const float* __restrict__ bb_w1, const float* __restrict__ bb_b1,
    const float* __restrict__ bb_w2, const float* __restrict__ bb_b2,
    const float* __restrict__ bb_g,  const float* __restrict__ bb_beta,
    const float* __restrict__ gU, const float* __restrict__ gV,
    const float* __restrict__ gb,
    const float* __restrict__ e_w1, const float* __restrict__ e_b1,
    const float* __restrict__ e_g,  const float* __restrict__ e_beta,
    const float* __restrict__ e_w2, const float* __restrict__ e_b2,
    const float* __restrict__ ut, float* __restrict__ out, int Btot)
{
  extern __shared__ char smem[];
  const int tid = threadIdx.x;

  _Float16* w1t  = (_Float16*)(smem + OFF_W1T);
  _Float16* w2t  = (_Float16*)(smem + OFF_W2T);
  _Float16* gUt  = (_Float16*)(smem + OFF_GUT);
  _Float16* gVt  = (_Float16*)(smem + OFF_GVT);
  _Float16* ew1t = (_Float16*)(smem + OFF_EW1T);
  _Float16* ew2t = (_Float16*)(smem + OFF_EW2T);
  float* bb_b1s = (float*)(smem + OFF_BB_B1);
  float* bb_b2s = (float*)(smem + OFF_BB_B2);
  float* bb_gs  = (float*)(smem + OFF_BB_G);
  float* bb_bes = (float*)(smem + OFF_BB_BE);
  float* gbs    = (float*)(smem + OFF_GB);
  float* e_b1s  = (float*)(smem + OFF_EB1);
  float* e_gs   = (float*)(smem + OFF_EG);
  float* e_bes  = (float*)(smem + OFF_EBE);
  float* e_b2s  = (float*)(smem + OFF_EB2);

  // ---- stage weights once per block: f32 -> f16, transposed [N][K], padded pitch ----
  for (int i = tid; i < 64*104; i += 256) { int n=i/104, k=i-n*104;
    w1t[i] = (_Float16)((k < 80) ? bb_w1[k*64 + n] : 0.f); }
  for (int i = tid; i < 64*72; i += 256)  { int n=i/72, k=i-n*72;
    w2t[i] = (_Float16)((k < 64) ? bb_w2[k*64 + n] : 0.f); }
  for (int i = tid; i < 128*72; i += 256) { int c=i/72, d=i-c*72;
    gUt[i] = (_Float16)((d < 64) ? gU[(c>>3)*512 + d*8 + (c&7)] : 0.f); }
  for (int i = tid; i < 128*40; i += 256) { int c=i/40, d=i-c*40;
    gVt[i] = (_Float16)((d < 16) ? gV[(c>>3)*128 + d*8 + (c&7)] : 0.f); }
  for (int i = tid; i < 16*64*72; i += 256) { int e=i/4608; int r=i-e*4608; int n=r/72, k=r-n*72;
    ew1t[i] = (_Float16)((k < 64) ? e_w1[e*4096 + k*64 + n] : 0.f); }
  for (int i = tid; i < 16*16*72; i += 256) { int e=i/1152; int r=i-e*1152; int c=r/72, k=r-c*72;
    ew2t[i] = (_Float16)((c < 10 && k < 64) ? e_w2[e*640 + k*10 + c] : 0.f); }
  for (int i = tid; i < 64; i += 256) {
    bb_b1s[i]=bb_b1[i]; bb_b2s[i]=bb_b2[i]; bb_gs[i]=bb_g[i]; bb_bes[i]=bb_beta[i]; }
  for (int i = tid; i < 16; i += 256) gbs[i] = gb[i];
  for (int i = tid; i < 1024; i += 256) { e_b1s[i]=e_b1[i]; e_gs[i]=e_g[i]; e_bes[i]=e_beta[i]; }
  for (int i = tid; i < 256; i += 256) { int e=i>>4, c=i&15;
    e_b2s[i] = (c < 10) ? e_b2[e*10 + c] : 0.f; }
  __syncthreads();

  const int wave = tid >> 5, lane = tid & 31;
  const int half = lane >> 4, col = lane & 15;
  const int tileBase = (blockIdx.x * NWAVES + wave) * 16;
  if (tileBase >= Btot) return;

  char* wb = smem + OFF_WAVE + wave * WAVE_BYTES;
  _Float16* xa   = (_Float16*)(wb);          // [16][96] f16 (phase 1)
  _Float16* ua   = (_Float16*)(wb);          // [16][32] f16 (phase 3, aliases xa)
  float*    gbuf = (float*)(wb + 1024);      // [16][16] f32
  _Float16* hst  = (_Float16*)(wb + 3072);   // [16][64] f16
  _Float16* zst  = (_Float16*)(wb + 5120);   // [16][64] f16
  float*    wbuf = (float*)(wb + 7168);      // [16][16] f32

  // ---- load x tile [16][80] -> f16 padded [16][96] ----
  for (int i = lane; i < 16*96; i += 32) { int r=i/96, j=i-r*96;
    xa[i] = (_Float16)((j < 80) ? x[(size_t)(tileBase + r) * 80 + j] : 0.f); }

  // ---- backbone fc1: [16,96]x[96,64], GELU -> hst ----
  v8f acc[4];
#pragma unroll
  for (int t = 0; t < 4; ++t) vzero(acc[t]);
#pragma unroll
  for (int kb = 0; kb < 3; ++kb) {
    v16h a  = ldfragA(xa, 96, kb * 32);
    v16h b0 = ldfragB(w1t + 0*16*104, 104, kb * 32);
    v16h b1 = ldfragB(w1t + 1*16*104, 104, kb * 32);
    v16h b2 = ldfragB(w1t + 2*16*104, 104, kb * 32);
    v16h b3 = ldfragB(w1t + 3*16*104, 104, kb * 32);
    acc[0] = WMMA_F16(a, b0, acc[0]);
    acc[1] = WMMA_F16(a, b1, acc[1]);
    acc[2] = WMMA_F16(a, b2, acc[2]);
    acc[3] = WMMA_F16(a, b3, acc[3]);
  }
#pragma unroll
  for (int t = 0; t < 4; ++t)
#pragma unroll
    for (int v = 0; v < 8; ++v) {
      int row = v + 8*half, c = t*16 + col;
      hst[row*64 + c] = (_Float16)gelu_f(acc[t][v] + bb_b1s[c]);
    }

  // ---- backbone fc2: [16,64]x[64,64], GELU, LayerNorm -> hst ----
#pragma unroll
  for (int t = 0; t < 4; ++t) vzero(acc[t]);
#pragma unroll
  for (int kb = 0; kb < 2; ++kb) {
    v16h a  = ldfragA(hst, 64, kb * 32);
    v16h b0 = ldfragB(w2t + 0*16*72, 72, kb * 32);
    v16h b1 = ldfragB(w2t + 1*16*72, 72, kb * 32);
    v16h b2 = ldfragB(w2t + 2*16*72, 72, kb * 32);
    v16h b3 = ldfragB(w2t + 3*16*72, 72, kb * 32);
    acc[0] = WMMA_F16(a, b0, acc[0]);
    acc[1] = WMMA_F16(a, b1, acc[1]);
    acc[2] = WMMA_F16(a, b2, acc[2]);
    acc[3] = WMMA_F16(a, b3, acc[3]);
  }
#pragma unroll
  for (int t = 0; t < 4; ++t)
#pragma unroll
    for (int v = 0; v < 8; ++v)
      acc[t][v] = gelu_f(acc[t][v] + bb_b2s[t*16 + col]);
  ln_store16(acc, bb_gs, bb_bes, hst);

  // ---- hoist normalized-h A fragments: reused by gate + all 16 experts ----
  v16h ha0 = ldfragA(hst, 64, 0);
  v16h ha1 = ldfragA(hst, 64, 32);

  // ---- user embedding tile [16][16] -> f16 padded [16][32] ----
  for (int i = lane; i < 16*32; i += 32) { int r=i>>5, j=i&31;
    float vv = 0.f;
    if (j < 16) { int u = uid[tileBase + r]; vv = ut[u*16 + j]; }
    ua[i] = (_Float16)vv; }
  v16h uaa = ldfragA(ua, 32, 0);

  // ---- bilinear gate: hU = h@gU', uV = u@gV', g = sum_r(hU*uV) ----
#pragma unroll
  for (int t = 0; t < 8; ++t) {
    v16h bu0 = ldfragB(gUt + t*16*72, 72, 0);
    v16h bu1 = ldfragB(gUt + t*16*72, 72, 32);
    v16h bv  = ldfragB(gVt + t*16*40, 40, 0);
    v8f hu; vzero(hu);
    hu = WMMA_F16(ha0, bu0, hu);
    hu = WMMA_F16(ha1, bu1, hu);
    v8f uv; vzero(uv);
    uv = WMMA_F16(uaa, bv, uv);
#pragma unroll
    for (int v = 0; v < 8; ++v) {
      float p = hu[v] * uv[v];
      p += __shfl_xor(p, 1, 32); p += __shfl_xor(p, 2, 32); p += __shfl_xor(p, 4, 32);
      if ((lane & 7) == 0)
        gbuf[(v + 8*half)*16 + (t*2 + ((lane >> 3) & 1))] = p;
    }
  }

  // ---- top-2 softmax over 16 experts (one lane per sample row) ----
  if (lane < 16) {
    float gv[16];
#pragma unroll
    for (int e = 0; e < 16; ++e) gv[e] = gbuf[lane*16 + e] + gbs[e];
    float m1 = -1e30f, m2 = -1e30f; int i1 = 0, i2 = 0;
#pragma unroll
    for (int e = 0; e < 16; ++e) {
      float g0 = gv[e];
      if (g0 > m1)      { m2 = m1; i2 = i1; m1 = g0; i1 = e; }
      else if (g0 > m2) { m2 = g0; i2 = e; }
    }
    float p2 = __expf(m2 - m1);
    float inv = __builtin_amdgcn_rcpf(1.f + p2);
#pragma unroll
    for (int e = 0; e < 16; ++e) wbuf[lane*16 + e] = 0.f;
    wbuf[lane*16 + i1] = inv;
    wbuf[lane*16 + i2] = p2 * inv;
  }

  // ---- experts: z = LN(gelu(h@W1+b1)); lpe = z@W2+b2; out += w_e * lpe ----
  v8f outacc; vzero(outacc);
  for (int e = 0; e < 16; ++e) {
    const _Float16* w1e = ew1t + e * 64 * 72;
#pragma unroll
    for (int t = 0; t < 4; ++t) vzero(acc[t]);
    {
      v16h b0 = ldfragB(w1e + 0*16*72, 72, 0);
      v16h b1 = ldfragB(w1e + 1*16*72, 72, 0);
      v16h b2 = ldfragB(w1e + 2*16*72, 72, 0);
      v16h b3 = ldfragB(w1e + 3*16*72, 72, 0);
      acc[0] = WMMA_F16(ha0, b0, acc[0]);
      acc[1] = WMMA_F16(ha0, b1, acc[1]);
      acc[2] = WMMA_F16(ha0, b2, acc[2]);
      acc[3] = WMMA_F16(ha0, b3, acc[3]);
    }
    {
      v16h b0 = ldfragB(w1e + 0*16*72, 72, 32);
      v16h b1 = ldfragB(w1e + 1*16*72, 72, 32);
      v16h b2 = ldfragB(w1e + 2*16*72, 72, 32);
      v16h b3 = ldfragB(w1e + 3*16*72, 72, 32);
      acc[0] = WMMA_F16(ha1, b0, acc[0]);
      acc[1] = WMMA_F16(ha1, b1, acc[1]);
      acc[2] = WMMA_F16(ha1, b2, acc[2]);
      acc[3] = WMMA_F16(ha1, b3, acc[3]);
    }
    const float* eb1 = e_b1s + e * 64;
#pragma unroll
    for (int t = 0; t < 4; ++t)
#pragma unroll
      for (int v = 0; v < 8; ++v)
        acc[t][v] = gelu_f(acc[t][v] + eb1[t*16 + col]);
    ln_store16(acc, e_gs + e*64, e_bes + e*64, zst);

    v16h za0 = ldfragA(zst, 64, 0);
    v16h za1 = ldfragA(zst, 64, 32);
    v16h c0  = ldfragB(ew2t + e*16*72, 72, 0);
    v16h c1  = ldfragB(ew2t + e*16*72, 72, 32);
    v8f lp; vzero(lp);
    lp = WMMA_F16(za0, c0, lp);
    lp = WMMA_F16(za1, c1, lp);
#pragma unroll
    for (int v = 0; v < 8; ++v) {
      int row = v + 8*half;
      float we = wbuf[row*16 + e];
      outacc[v] += we * (lp[v] + e_b2s[e*16 + col]);
    }
  }

  // ---- store logits [16][10] ----
  if (col < 10) {
#pragma unroll
    for (int v = 0; v < 8; ++v) {
      int row = v + 8*half;
      out[(size_t)(tileBase + row) * 10 + col] = outacc[v];
    }
  }
}

extern "C" void kernel_launch(void* const* d_in, const int* in_sizes, int n_in,
                              void* d_out, int out_size, void* d_ws, size_t ws_size,
                              hipStream_t stream) {
  (void)n_in; (void)out_size; (void)d_ws; (void)ws_size;
  const float* x       = (const float*)d_in[0];
  const int*   uid     = (const int*)  d_in[1];
  const float* bb_w1   = (const float*)d_in[2];
  const float* bb_b1   = (const float*)d_in[3];
  const float* bb_w2   = (const float*)d_in[4];
  const float* bb_b2   = (const float*)d_in[5];
  const float* bb_g    = (const float*)d_in[6];
  const float* bb_beta = (const float*)d_in[7];
  const float* gU      = (const float*)d_in[8];
  const float* gV      = (const float*)d_in[9];
  const float* gb      = (const float*)d_in[10];
  const float* e_w1    = (const float*)d_in[11];
  const float* e_b1    = (const float*)d_in[12];
  const float* e_g     = (const float*)d_in[13];
  const float* e_beta  = (const float*)d_in[14];
  const float* e_w2    = (const float*)d_in[15];
  const float* e_b2    = (const float*)d_in[16];
  const float* ut      = (const float*)d_in[17];
  float* out = (float*)d_out;

  const int Btot = in_sizes[1];                     // user_ids count == batch
  const int blocks = (Btot + NWAVES*16 - 1) / (NWAVES*16);

  (void)hipFuncSetAttribute((const void*)moe_fused_kernel,
                            hipFuncAttributeMaxDynamicSharedMemorySize, LDS_TOTAL);

  moe_fused_kernel<<<blocks, 256, LDS_TOTAL, stream>>>(
      x, uid, bb_w1, bb_b1, bb_w2, bb_b2, bb_g, bb_beta,
      gU, gV, gb, e_w1, e_b1, e_g, e_beta, e_w2, e_b2, ut, out, Btot);
}
